// Edgewise_88905823027830
// MI455X (gfx1250) — matrix operand order repaired and checked by reading, not compiled, not run on hardware
//
#include <hip/hip_runtime.h>

// ---------------- types / helpers ----------------
typedef __attribute__((ext_vector_type(16))) __bf16 v16bf;
typedef __attribute__((ext_vector_type(8)))  float  v8f;

union FragBF { v16bf v; uint4 q[2]; };

__device__ __forceinline__ unsigned short f2bf(float f) {
  unsigned int u = __float_as_uint(f);
  u += 0x7FFFu + ((u >> 16) & 1u);           // round-to-nearest-even
  return (unsigned short)(u >> 16);
}
__device__ __forceinline__ float bf2f(unsigned short h) {
  return __uint_as_float(((unsigned int)h) << 16);
}
__device__ __forceinline__ float sigmoidf_(float x) { return 1.f / (1.f + __expf(-x)); }

__device__ __forceinline__ v8f wmma_bf16(const FragBF& a, const FragBF& b, v8f c) {
  return __builtin_amdgcn_wmma_f32_16x16x32_bf16(false, a.v, false, b.v, (short)0, c, false, false);
}

// async copy 16B global -> LDS, tracked by ASYNCcnt (CDNA5)
#define ASYNC_B128(ldsoff, src) \
  asm volatile("global_load_async_to_lds_b128 %0, %1, off" :: "v"(ldsoff), "v"(src) : "memory")

// ---------------- weight f32 -> bf16 conversion ----------------
__global__ void cvt_bf16_kernel(const float* __restrict__ s, unsigned short* __restrict__ d, int n) {
  int i = blockIdx.x * 256 + threadIdx.x;
  if (i < n) d[i] = f2bf(s[i]);
}

// ---------------- radial MLP layer 1 + LayerNorm + SiLU ----------------
__global__ void radial1_kernel(const float* __restrict__ xe,
                               const float* __restrict__ w1, const float* __restrict__ b1,
                               const float* __restrict__ lng, const float* __restrict__ lnb,
                               unsigned short* __restrict__ hbuf) {
  __shared__ float xs[128];
  __shared__ float red[64];
  int e = blockIdx.x, o = threadIdx.x;
  xs[o]      = xe[(long)e * 128 + o];
  xs[o + 64] = xe[(long)e * 128 + 64 + o];
  __syncthreads();
  const float* wr = w1 + o * 128;
  float h = b1[o];
#pragma unroll 8
  for (int i = 0; i < 128; ++i) h = fmaf(wr[i], xs[i], h);
  red[o] = h; __syncthreads();
  for (int s = 32; s > 0; s >>= 1) { if (o < s) red[o] += red[o + s]; __syncthreads(); }
  float mu = red[0] * (1.f / 64.f);
  __syncthreads();
  float d = h - mu;
  red[o] = d * d; __syncthreads();
  for (int s = 32; s > 0; s >>= 1) { if (o < s) red[o] += red[o + s]; __syncthreads(); }
  float var = red[0] * (1.f / 64.f);
  float y = d * rsqrtf(var + 1e-5f) * lng[o] + lnb[o];
  float sv = y * sigmoidf_(y);
  hbuf[(long)e * 64 + o] = f2bf(sv);
}

// ---------------- gather + wigner rotate + radial scale -> bf16 A matrix ----------------
__global__ void wigner_gather_kernel(const float* __restrict__ x,
                                     const float* __restrict__ wigner,
                                     const unsigned short* __restrict__ xrad,
                                     const int* __restrict__ eidx,
                                     unsigned short* __restrict__ abuf, int E) {
  __shared__ float Wl[19 * 25];
  int e = blockIdx.x;
  int c = threadIdx.x;
  for (int i = c; i < 475; i += 256) Wl[i] = wigner[(long)e * 475 + i];
  __syncthreads();
  int src = eidx[e], dst = eidx[E + e];
  const float* xp = x + (long)(c < 128 ? src : dst) * 3200 + (c & 127);
  float xv[25];
#pragma unroll
  for (int n = 0; n < 25; ++n) xv[n] = xp[n * 128];
  const unsigned short* rp = xrad + (long)e * 3072;
  unsigned short* ap = abuf + (long)e * 4864 + c;
  const int radoff[19] = {0,256,512,768,1024, 1280,1536,1792,2048, 1280,1536,1792,2048,
                          2304,2560,2816, 2304,2560,2816};
#pragma unroll
  for (int row = 0; row < 19; ++row) {
    float acc = 0.f;
#pragma unroll
    for (int n = 0; n < 25; ++n) acc = fmaf(Wl[row * 25 + n], xv[n], acc);
    float rv = bf2f(rp[radoff[row] + c]);
    ap[row * 256] = f2bf(acc * rv);
  }
}

// ---------------- bf16 WMMA GEMM with async-LDS double buffering ----------------
// C(R,N) = A(R,K) * Bw(N,K)^T + bias.  Block tile 64(M) x 128(N), K-step 64.
// 8 waves as 2x4 grid of 32x32 macro-tiles; each wave holds 4 f32 16x16 accums
// and issues 8 v_wmma per K-step. A/B tiles staged with
// global_load_async_to_lds_b128 (ASYNCcnt) into double-buffered LDS.
__global__ void __launch_bounds__(256)
gemm_bf16_kernel(const unsigned short* __restrict__ A,
                 long abase, int aestride, int ashift, int aspan,
                 const unsigned short* __restrict__ Bw,
                 const float* __restrict__ bias,
                 void* __restrict__ Cout, int cbf, int N, int K) {
  __shared__ __align__(16) unsigned short As[2][64 * 64];     // 2 x 8 KB
  __shared__ __align__(16) unsigned short Bs[2][128 * 64];    // 2 x 16 KB
  const int tid = threadIdx.x;
  const int wave = tid >> 5, lane = tid & 31;
  const int m0 = blockIdx.x << 6;
  const int n0 = blockIdx.y << 7;
  const int wm = (wave >> 2) << 5;          // 0 / 32
  const int wn = (wave & 3) << 5;           // 0 / 32 / 64 / 96
  const int amask = (1 << ashift) - 1;

  // per-thread async-copy chunks: A = 2 x 16B, B = 4 x 16B per buffer
  const char* asrc[2];  unsigned alds[2];
#pragma unroll
  for (int c = 0; c < 2; ++c) {
    int ci = tid + (c << 8);                // 0..511 -> row = ci>>3, koff = (ci&7)*8
    int row = ci >> 3, ko = (ci & 7) << 3;
    int r = m0 + row;
    long aoff = (long)(r >> ashift) * aestride + abase + (long)(r & amask) * aspan;
    asrc[c] = (const char*)(A + aoff + ko);
    alds[c] = (unsigned)(size_t)(&As[0][0]) + ((unsigned)ci << 4);
  }
  const char* bsrc[4];  unsigned blds[4];
#pragma unroll
  for (int c = 0; c < 4; ++c) {
    int ci = tid + (c << 8);                // 0..1023
    int row = ci >> 3, ko = (ci & 7) << 3;
    bsrc[c] = (const char*)(Bw + (long)(n0 + row) * K + ko);
    blds[c] = (unsigned)(size_t)(&Bs[0][0]) + ((unsigned)ci << 4);
  }

  v8f acc00 = {0.f,0.f,0.f,0.f,0.f,0.f,0.f,0.f};
  v8f acc01 = acc00, acc10 = acc00, acc11 = acc00;

  // fragment lane addressing (ISA 16-bit A 16x32 / B 32x16 layouts)
  const int am0 = wm + (lane & 15);
  const int akb = (lane >> 4) << 3;         // 0 or 8
  const int bn0 = wn + (lane & 15);
  const int bkb = (lane >> 4) << 3;         // 0 or 8

  // prologue: stage buffer 0
#pragma unroll
  for (int c = 0; c < 2; ++c) ASYNC_B128(alds[c], asrc[c]);
#pragma unroll
  for (int c = 0; c < 4; ++c) ASYNC_B128(blds[c], bsrc[c]);

  const int nk = K >> 6;
  for (int it = 0; it < nk; ++it) {
    const int buf = it & 1;
    if (it + 1 < nk) {
      const int kb = (it + 1) << 7;                       // byte advance: 64 elems
      const unsigned ao = (unsigned)((buf ^ 1) * (64 * 64 * 2));
      const unsigned bo = (unsigned)((buf ^ 1) * (128 * 64 * 2));
#pragma unroll
      for (int c = 0; c < 2; ++c) ASYNC_B128(alds[c] + ao, asrc[c] + kb);
#pragma unroll
      for (int c = 0; c < 4; ++c) ASYNC_B128(blds[c] + bo, bsrc[c] + kb);
      asm volatile("s_wait_asynccnt 0x6" ::: "memory");   // current buffer done
    } else {
      asm volatile("s_wait_asynccnt 0x0" ::: "memory");
    }
    __syncthreads();

    const unsigned short* ab = &As[buf][0];
    const unsigned short* bb = &Bs[buf][0];
#pragma unroll
    for (int kc = 0; kc < 2; ++kc) {
      FragBF a0, a1, b0, b1;
      const unsigned short* ap0 = ab + am0 * 64 + kc * 32 + akb;
      a0.q[0] = *(const uint4*)ap0;        a0.q[1] = *(const uint4*)(ap0 + 16);
      const unsigned short* ap1 = ap0 + 16 * 64;
      a1.q[0] = *(const uint4*)ap1;        a1.q[1] = *(const uint4*)(ap1 + 16);
      const unsigned short* bp0 = bb + bn0 * 64 + kc * 32 + bkb;
      b0.q[0] = *(const uint4*)bp0;        b0.q[1] = *(const uint4*)(bp0 + 16);
      const unsigned short* bp1 = bp0 + 16 * 64;
      b1.q[0] = *(const uint4*)bp1;        b1.q[1] = *(const uint4*)(bp1 + 16);
      acc00 = wmma_bf16(a0, b0, acc00);
      acc01 = wmma_bf16(a0, b1, acc01);
      acc10 = wmma_bf16(a1, b0, acc10);
      acc11 = wmma_bf16(a1, b1, acc11);
    }
    __syncthreads();
  }

  // store: D layout -> m = r + 8*(lane>=16), n = lane&15
  v8f* accs[4] = {&acc00, &acc01, &acc10, &acc11};
#pragma unroll
  for (int mi = 0; mi < 2; ++mi) {
#pragma unroll
    for (int ni = 0; ni < 2; ++ni) {
      v8f a = *accs[mi * 2 + ni];
      int cn = n0 + wn + ni * 16 + (lane & 15);
      int mb = m0 + wm + mi * 16 + ((lane >> 4) << 3);
      float bv = bias ? bias[cn] : 0.f;
      if (cbf) {
        unsigned short* C = (unsigned short*)Cout;
#pragma unroll
        for (int rr = 0; rr < 8; ++rr) C[(long)(mb + rr) * N + cn] = f2bf(a[rr] + bv);
      } else {
        float* C = (float*)Cout;
#pragma unroll
        for (int rr = 0; rr < 8; ++rr) C[(long)(mb + rr) * N + cn] = a[rr] + bv;
      }
    }
  }
}

// ---------------- conv1 epilogue: complex recombine + sigmoid gates + SiLU ----------------
__global__ void epilogue1_kernel(const float* __restrict__ Y0, const float* __restrict__ Y1,
                                 const float* __restrict__ Y2, unsigned short* __restrict__ H) {
  int e = blockIdx.x, c = threadIdx.x;
  const float* y0 = Y0 + (long)e * 1152;
  unsigned short* hp = H + (long)e * 2432;
  float g[4];
#pragma unroll
  for (int l = 0; l < 4; ++l) g[l] = sigmoidf_(y0[l * 128 + c]);
  float v0 = y0[512 + c];
  hp[c] = f2bf(v0 * sigmoidf_(v0));                        // row 0: SiLU
#pragma unroll
  for (int l = 1; l < 5; ++l) {                            // rows 1..4
    float v = y0[512 + l * 128 + c];
    hp[l * 128 + c] = f2bf(v * g[l - 1]);
  }
  const float* ya = Y1 + (long)(2 * e) * 1024;
  const float* yb = ya + 1024;
#pragma unroll
  for (int l = 0; l < 4; ++l) {                            // m=1: rows 5..8, 9..12
    int o = l * 128 + c;
    float h0 = ya[o] - yb[512 + o];
    float h1 = yb[o] + ya[512 + o];
    hp[(5 + l) * 128 + c] = f2bf(h0 * g[l]);
    hp[(9 + l) * 128 + c] = f2bf(h1 * g[l]);
  }
  const float* za = Y2 + (long)(2 * e) * 768;
  const float* zb = za + 768;
#pragma unroll
  for (int l = 0; l < 3; ++l) {                            // m=2: rows 13..15, 16..18
    int o = l * 128 + c;
    float h0 = za[o] - zb[384 + o];
    float h1 = zb[o] + za[384 + o];
    hp[(13 + l) * 128 + c] = f2bf(h0 * g[l + 1]);
    hp[(16 + l) * 128 + c] = f2bf(h1 * g[l + 1]);
  }
}

// ---------------- conv2 epilogue: complex recombine + envelope -> bf16 ----------------
__global__ void epilogue2_kernel(const float* __restrict__ Z0, const float* __restrict__ Z1,
                                 const float* __restrict__ Z2, const float* __restrict__ env,
                                 unsigned short* __restrict__ Z) {
  int e = blockIdx.x, c = threadIdx.x;
  float ev = env[e];
  unsigned short* zp = Z + (long)e * 2432;
  const float* z0 = Z0 + (long)e * 640;
#pragma unroll
  for (int l = 0; l < 5; ++l) zp[l * 128 + c] = f2bf(z0[l * 128 + c] * ev);
  const float* ya = Z1 + (long)(2 * e) * 1024;
  const float* yb = ya + 1024;
#pragma unroll
  for (int l = 0; l < 4; ++l) {
    int o = l * 128 + c;
    zp[(5 + l) * 128 + c] = f2bf((ya[o] - yb[512 + o]) * ev);
    zp[(9 + l) * 128 + c] = f2bf((yb[o] + ya[512 + o]) * ev);
  }
  const float* za = Z2 + (long)(2 * e) * 768;
  const float* zb = za + 768;
#pragma unroll
  for (int l = 0; l < 3; ++l) {
    int o = l * 128 + c;
    zp[(13 + l) * 128 + c] = f2bf((za[o] - zb[384 + o]) * ev);
    zp[(16 + l) * 128 + c] = f2bf((zb[o] + za[384 + o]) * ev);
  }
}

// ---------------- inverse wigner + atomic scatter into node accumulator ----------------
__global__ void winv_scatter_kernel(const float* __restrict__ winv,
                                    const unsigned short* __restrict__ zbuf,
                                    const int* __restrict__ eidx,
                                    float* __restrict__ out, int E) {
  __shared__ float Wl[25 * 19];
  int e = blockIdx.x, t = threadIdx.x;
  for (int i = t; i < 475; i += 256) Wl[i] = winv[(long)e * 475 + i];
  __syncthreads();
  int c = t & 127, sub = t >> 7;
  float zv[19];
  const unsigned short* zp = zbuf + (long)e * 2432 + c;
#pragma unroll
  for (int k = 0; k < 19; ++k) zv[k] = bf2f(zp[k * 128]);
  int dst = eidx[E + e];
  float* op = out + (long)dst * 3200 + c;
  for (int n = sub; n < 25; n += 2) {
    float acc = 0.f;
#pragma unroll
    for (int k = 0; k < 19; ++k) acc = fmaf(Wl[n * 19 + k], zv[k], acc);
    atomicAdd(op + n * 128, acc);
  }
}

// ---------------- host launch ----------------
extern "C" void kernel_launch(void* const* d_in, const int* in_sizes, int n_in,
                              void* d_out, int out_size, void* d_ws, size_t ws_size,
                              hipStream_t stream) {
  (void)in_sizes; (void)n_in; (void)out_size; (void)ws_size;
  const int E = 20000, NODES = 1600;
  const int EP = 20032;                 // E padded to a multiple of 64 (GEMM M-tile)

  const float* x      = (const float*)d_in[0];
  const float* x_edge = (const float*)d_in[1];
  const float* env    = (const float*)d_in[2];
  const float* wig    = (const float*)d_in[3];
  const float* wigi   = (const float*)d_in[4];
  const float* rad_w1 = (const float*)d_in[5];
  const float* rad_b1 = (const float*)d_in[6];
  const float* lng    = (const float*)d_in[7];
  const float* lnb    = (const float*)d_in[8];
  const float* rad_w2 = (const float*)d_in[9];
  const float* rad_b2 = (const float*)d_in[10];
  const float* c1w0   = (const float*)d_in[11];
  const float* c1b0   = (const float*)d_in[12];
  const float* c1w1   = (const float*)d_in[13];
  const float* c1w2   = (const float*)d_in[14];
  const float* c2w0   = (const float*)d_in[15];
  const float* c2b0   = (const float*)d_in[16];
  const float* c2w1   = (const float*)d_in[17];
  const float* c2w2   = (const float*)d_in[18];
  const int*   eidx   = (const int*)d_in[19];

  char* ws = (char*)d_ws;
  size_t off = 0;
  auto take = [&](size_t bytes) -> char* {
    char* p = ws + off;
    off += (bytes + 255) & ~(size_t)255;
    return p;
  };

  // bf16 weight copies
  unsigned short* wbR2  = (unsigned short*)take((size_t)3072 * 64 * 2);
  unsigned short* wbC10 = (unsigned short*)take((size_t)1152 * 1280 * 2);
  unsigned short* wbC11 = (unsigned short*)take((size_t)1024 * 1024 * 2);
  unsigned short* wbC12 = (unsigned short*)take((size_t)768 * 768 * 2);
  unsigned short* wbC20 = (unsigned short*)take((size_t)640 * 640 * 2);
  unsigned short* wbC21 = (unsigned short*)take((size_t)1024 * 512 * 2);
  unsigned short* wbC22 = (unsigned short*)take((size_t)768 * 384 * 2);
  // activations (row-padded to EP where they are a GEMM A or C operand)
  unsigned short* hbuf  = (unsigned short*)take((size_t)EP * 64 * 2);
  unsigned short* xrad  = (unsigned short*)take((size_t)EP * 3072 * 2);  // later reused as zbuf
  unsigned short* abuf  = (unsigned short*)take((size_t)EP * 4864 * 2);
  char* yreg = take((size_t)EP * 1152 * 4 + (size_t)2 * E * 1024 * 4 + (size_t)2 * E * 768 * 4);
  float* y0 = (float*)yreg;
  float* y1 = y0 + (size_t)EP * 1152;
  float* y2 = y1 + (size_t)2 * E * 1024;
  unsigned short* hmsg = (unsigned short*)take((size_t)EP * 2432 * 2);
  // aliased buffers (producers fully consumed before these are written)
  float* z0 = (float*)yreg;                 // over Y region (EP*640 + 2E*1024 + 2E*768 fits)
  float* z1 = z0 + (size_t)EP * 640;
  float* z2 = z1 + (size_t)2 * E * 1024;
  unsigned short* zbuf = xrad;              // over x_rad region (E*2432 <= EP*3072)

  auto cvt = [&](const float* s, unsigned short* d, int n) {
    cvt_bf16_kernel<<<(n + 255) / 256, 256, 0, stream>>>(s, d, n);
  };
  cvt(rad_w2, wbR2, 3072 * 64);
  cvt(c1w0, wbC10, 1152 * 1280);
  cvt(c1w1, wbC11, 1024 * 1024);
  cvt(c1w2, wbC12, 768 * 768);
  cvt(c2w0, wbC20, 640 * 640);
  cvt(c2w1, wbC21, 1024 * 512);
  cvt(c2w2, wbC22, 768 * 384);

  // radial MLP
  radial1_kernel<<<E, 64, 0, stream>>>(x_edge, rad_w1, rad_b1, lng, lnb, hbuf);
  gemm_bf16_kernel<<<dim3(EP / 64, 3072 / 128), 256, 0, stream>>>(
      hbuf, 0, 64, 0, 0, wbR2, rad_b2, xrad, 1, 3072, 64);

  // gather + wigner + radial scaling
  wigner_gather_kernel<<<E, 256, 0, stream>>>(x, wig, xrad, eidx, abuf, E);

  // SO2 conv 1
  gemm_bf16_kernel<<<dim3(EP / 64, 1152 / 128), 256, 0, stream>>>(
      abuf, 0, 4864, 0, 0, wbC10, c1b0, y0, 0, 1152, 1280);
  gemm_bf16_kernel<<<dim3(2 * E / 64, 1024 / 128), 256, 0, stream>>>(
      abuf, 1280, 4864, 1, 1024, wbC11, nullptr, y1, 0, 1024, 1024);
  gemm_bf16_kernel<<<dim3(2 * E / 64, 768 / 128), 256, 0, stream>>>(
      abuf, 3328, 4864, 1, 768, wbC12, nullptr, y2, 0, 768, 768);
  epilogue1_kernel<<<E, 128, 0, stream>>>(y0, y1, y2, hmsg);

  // SO2 conv 2
  gemm_bf16_kernel<<<dim3(EP / 64, 640 / 128), 256, 0, stream>>>(
      hmsg, 0, 2432, 0, 0, wbC20, c2b0, z0, 0, 640, 640);
  gemm_bf16_kernel<<<dim3(2 * E / 64, 1024 / 128), 256, 0, stream>>>(
      hmsg, 640, 2432, 1, 512, wbC21, nullptr, z1, 0, 1024, 512);
  gemm_bf16_kernel<<<dim3(2 * E / 64, 768 / 128), 256, 0, stream>>>(
      hmsg, 1664, 2432, 1, 384, wbC22, nullptr, z2, 0, 768, 384);
  epilogue2_kernel<<<E, 128, 0, stream>>>(z0, z1, z2, env, zbuf);

  // inverse wigner + scatter
  hipMemsetAsync(d_out, 0, (size_t)NODES * 3200 * 4, stream);
  winv_scatter_kernel<<<E, 256, 0, stream>>>(wigi, zbuf, eidx, (float*)d_out, E);
}
